// EncoderBlock_5188320494088
// MI455X (gfx1250) — matrix operand order repaired and checked by reading, not compiled, not run on hardware
//
#include <hip/hip_runtime.h>
#include <hip/hip_bf16.h>
#include <math.h>

// ---------------------------------------------------------------------------
// Transformer encoder block for MI455X (gfx1250, wave32, WMMA + TDM).
// B=8 L=1024 E=1024 H=16 DH=64 FF=4096.  fp16 data / fp32 accumulate WMMA.
// ---------------------------------------------------------------------------

#define B_  8
#define L_  1024
#define E_  1024
#define H_  16
#define DH_ 64
#define FF_ 4096

typedef __attribute__((ext_vector_type(16))) _Float16 v16h;
typedef __attribute__((ext_vector_type(8)))  _Float16 v8h;
typedef __attribute__((ext_vector_type(8)))  float    v8f;
typedef __attribute__((ext_vector_type(4)))  unsigned v4u;
typedef __attribute__((ext_vector_type(8)))  int      v8i;
typedef __attribute__((ext_vector_type(4)))  int      v4i;

union V16 { v16h v; v8h h[2]; };

#if defined(__HIP_DEVICE_COMPILE__) && \
    __has_builtin(__builtin_amdgcn_tensor_load_to_lds) && \
    __has_builtin(__builtin_amdgcn_s_wait_tensorcnt)
#define USE_TDM 1
#else
#define USE_TDM 0
#endif

#if defined(__HIP_DEVICE_COMPILE__) && __has_builtin(__builtin_amdgcn_permlane16)
#define USE_PERMLANE 1
#else
#define USE_PERMLANE 0
#endif

__device__ __forceinline__ v8f wmma_f16(v16h a, v16h b, v8f c) {
  return __builtin_amdgcn_wmma_f32_16x16x32_f16(
      false, a, false, b, (short)0, c, false, false);
}

#if USE_TDM
// 2D tile load via Tensor Data Mover.  Units of 8 bytes for dims/strides.
// pad_int_code: dwords-before-pad = 2<<code ; pad_amt_code: pad dwords = code+1.
__device__ __forceinline__ void tdm_load_2d(unsigned lds_off, const void* gptr,
                                            unsigned row_u8, unsigned nrows,
                                            unsigned stride_u8,
                                            unsigned pad_int_code,
                                            unsigned pad_amt_code) {
  unsigned long long ga = (unsigned long long)(size_t)gptr;
  v4u g0 = {1u,                                   // count=1 valid descriptor
            lds_off,                              // lds_addr
            (unsigned)ga,                         // global_addr[31:0]
            ((unsigned)(ga >> 32) & 0x01FFFFFFu) | 0x80000000u};  // addr+type=2
  v8i g1 = {(int)((3u << 16) |                    // data_size = 8B
                  (1u << 20) |                    // pad_enable
                  (pad_int_code << 22) | (pad_amt_code << 25)),
            (int)(row_u8 << 16),                  // tensor_dim0 (low 16)
            (int)(nrows << 16),                   // tensor_dim1 (low 16)
            (int)(row_u8 << 16),                  // tile_dim0
            (int)nrows,                           // tile_dim1 (tile_dim2=0)
            (int)stride_u8,                       // tensor_dim0_stride low32
            0, 0};
  v4i z4 = {0, 0, 0, 0};
  v8i z8 = {0, 0, 0, 0, 0, 0, 0, 0};
  __builtin_amdgcn_tensor_load_to_lds(g0, g1, z4, z4, z8, 0);
}
#endif

// max-reduce across the 16 lanes of each wave half (row of a WMMA C-frag)
__device__ __forceinline__ float rowmax16(float v) {
#if USE_PERMLANE
  {
    unsigned u, r;
    u = __builtin_bit_cast(unsigned, v);
    r = __builtin_amdgcn_permlane16(u, u, 0x67452301u, 0xEFCDAB89u, false, false);
    v = fmaxf(v, __builtin_bit_cast(float, r));   // xor 1
    u = __builtin_bit_cast(unsigned, v);
    r = __builtin_amdgcn_permlane16(u, u, 0x54761032u, 0xDCFE98BAu, false, false);
    v = fmaxf(v, __builtin_bit_cast(float, r));   // xor 2
    u = __builtin_bit_cast(unsigned, v);
    r = __builtin_amdgcn_permlane16(u, u, 0x32107654u, 0xBA98FEDCu, false, false);
    v = fmaxf(v, __builtin_bit_cast(float, r));   // xor 4
    u = __builtin_bit_cast(unsigned, v);
    r = __builtin_amdgcn_permlane16(u, u, 0xFEDCBA98u, 0x76543210u, false, false);
    v = fmaxf(v, __builtin_bit_cast(float, r));   // xor 8
  }
#else
#pragma unroll
  for (int o = 1; o < 16; o <<= 1) v = fmaxf(v, __shfl_xor(v, o, 32));
#endif
  return v;
}

// ---------------------------------------------------------------------------
// fp32 -> fp16 conversions (plain, and transposing [batch,K,N]->[batch,N,K])
// ---------------------------------------------------------------------------
__global__ __launch_bounds__(256) void cvt_kernel(const float* __restrict__ in,
                                                  _Float16* __restrict__ out,
                                                  int n) {
  int i = blockIdx.x * 256 + threadIdx.x;
  if (i < n) out[i] = (_Float16)in[i];
}

__global__ __launch_bounds__(256) void cvt_tr_kernel(
    const float* __restrict__ in, _Float16* __restrict__ out, int Kd, int Nd) {
  const size_t bofs = (size_t)blockIdx.y * Kd * Nd;
  int i = blockIdx.x * 256 + threadIdx.x;
  if (i < Kd * Nd) {
    int k = i / Nd, n = i % Nd;
    out[bofs + (size_t)n * Kd + k] = (_Float16)in[bofs + i];
  }
}

// ---------------------------------------------------------------------------
// WMMA GEMM: C[M,N] = A[M,K] * Wt[N,K]^T  (both operands row-major-contiguous
// fp16, K-contiguous -> TDM-friendly).  Block tile 128x128, BK=32, double-
// buffered LDS filled by the Tensor Data Mover; 8 waves, each 32x64 (2x4 wmma).
// EPI: 0 = QK proj (out fp16 scattered to [B,H,L,DH])
//      1 = OUTPROJ (out fp32 = acc + bias + resid)
//      2 = FFN1    (out fp16 = relu(acc + bias))
//      3 = FFN2    (out fp32 = acc + bias + resid)
//      4 = V proj  (out fp16 scattered to [B,H,DH,L] -- pre-transposed for TDM)
// ---------------------------------------------------------------------------
template <int EPI>
__global__ __launch_bounds__(256) void gemm_kernel(
    const _Float16* __restrict__ A, int lda, int K,
    const _Float16* __restrict__ Wt, int N,
    const float* __restrict__ bias,
    const float* __restrict__ resid,
    void* __restrict__ outp) {
  constexpr int TS = 40;  // LDS row stride in fp16 (80B: 16B-aligned, skewed)
  __shared__ _Float16 As[2][128 * TS];
  __shared__ _Float16 Bs[2][128 * TS];

  const int tid  = threadIdx.x;
  const int wave = tid >> 5, lane = tid & 31;
  const int l15  = lane & 15, lh = lane >> 4;
  const int wm   = wave & 3,  wn = wave >> 2;
  const int m0   = blockIdx.y * 128, n0 = blockIdx.x * 128;

  const v8f vzero = {0.f, 0.f, 0.f, 0.f, 0.f, 0.f, 0.f, 0.f};
  v8f c[2][4];
#pragma unroll
  for (int i = 0; i < 2; ++i)
#pragma unroll
    for (int j = 0; j < 4; ++j) c[i][j] = vzero;

  const _Float16* abase = A + (size_t)m0 * lda;
  const _Float16* bbase = Wt + (size_t)n0 * (size_t)K;
#if USE_TDM
  const unsigned asOff[2] = {(unsigned)(size_t)(const void*)&As[0][0],
                             (unsigned)(size_t)(const void*)&As[1][0]};
  const unsigned bsOff[2] = {(unsigned)(size_t)(const void*)&Bs[0][0],
                             (unsigned)(size_t)(const void*)&Bs[1][0]};
  // Prologue: wave 0 kicks the TDM for tile 0 (row=64B=16dw -> pad code 3;
  // pad 16B=4dw -> code 3).
  if (wave == 0) {
    tdm_load_2d(asOff[0], abase, 8, 128, (unsigned)(lda >> 2), 3, 3);
    tdm_load_2d(bsOff[0], bbase, 8, 128, (unsigned)(K >> 2), 3, 3);
  }
#else
  const int crow = tid >> 1, ckc = (tid & 1) * 16;  // cooperative-copy fallback
  {
    const _Float16* sa = abase + (size_t)crow * lda + ckc;
    *(v8h*)&As[0][crow * TS + ckc]     = *(const v8h*)(sa);
    *(v8h*)&As[0][crow * TS + ckc + 8] = *(const v8h*)(sa + 8);
    const _Float16* sb = bbase + (size_t)crow * K + ckc;
    *(v8h*)&Bs[0][crow * TS + ckc]     = *(const v8h*)(sb);
    *(v8h*)&Bs[0][crow * TS + ckc + 8] = *(const v8h*)(sb + 8);
  }
#endif

  const int nk = K >> 5;
  for (int i = 0; i < nk; ++i) {
    const int buf = i & 1;
    __syncthreads();  // all waves finished reading buf^1 (previous tile)
#if USE_TDM
    if (wave == 0) {
      if (i + 1 < nk) {  // overlap: fetch tile i+1 while we compute tile i
        const int k1 = (i + 1) << 5;
        tdm_load_2d(asOff[buf ^ 1], abase + k1, 8, 128, (unsigned)(lda >> 2), 3, 3);
        tdm_load_2d(bsOff[buf ^ 1], bbase + k1, 8, 128, (unsigned)(K >> 2), 3, 3);
        __builtin_amdgcn_s_wait_tensorcnt(2);  // oldest two (tile i) done
      } else {
        __builtin_amdgcn_s_wait_tensorcnt(0);
      }
    }
#else
    if (i + 1 < nk) {
      const int k1 = (i + 1) << 5;
      const _Float16* sa = abase + (size_t)crow * lda + k1 + ckc;
      *(v8h*)&As[buf ^ 1][crow * TS + ckc]     = *(const v8h*)(sa);
      *(v8h*)&As[buf ^ 1][crow * TS + ckc + 8] = *(const v8h*)(sa + 8);
      const _Float16* sb = bbase + (size_t)crow * K + k1 + ckc;
      *(v8h*)&Bs[buf ^ 1][crow * TS + ckc]     = *(const v8h*)(sb);
      *(v8h*)&Bs[buf ^ 1][crow * TS + ckc + 8] = *(const v8h*)(sb + 8);
    }
#endif
    __syncthreads();  // tile i visible to every wave

    V16 af[2];
#pragma unroll
    for (int mi = 0; mi < 2; ++mi) {  // A frag: row=l15, k=16*(e>=8)+8*lh+(e&7)
      const _Float16* p = &As[buf][(wm * 32 + mi * 16 + l15) * TS + 8 * lh];
      af[mi].h[0] = *(const v8h*)p;
      af[mi].h[1] = *(const v8h*)(p + 16);
    }
#pragma unroll
    for (int ni = 0; ni < 4; ++ni) {  // B frag: n=l15, k=16*(e>=8)+8*lh+(e&7)
      V16 bfr;
      const _Float16* p = &Bs[buf][(wn * 64 + ni * 16 + l15) * TS + 8 * lh];
      bfr.h[0] = *(const v8h*)p;
      bfr.h[1] = *(const v8h*)(p + 16);
#pragma unroll
      for (int mi = 0; mi < 2; ++mi)
        c[mi][ni] = wmma_f16(af[mi].v, bfr.v, c[mi][ni]);
    }
  }

  // Epilogue. C frag: VGPR r / lane l -> row = r + 8*(l>=16), col = l&15
#pragma unroll
  for (int mi = 0; mi < 2; ++mi)
#pragma unroll
    for (int ni = 0; ni < 4; ++ni)
#pragma unroll
      for (int r = 0; r < 8; ++r) {
        const int row = m0 + wm * 32 + mi * 16 + r + 8 * lh;
        const int col = n0 + wn * 64 + ni * 16 + l15;
        const float acc = c[mi][ni][r] + bias[col];
        if constexpr (EPI == 0) {
          const int bb = row >> 10, ll = row & (L_ - 1);
          const int hh = col >> 6, dd = col & 63;
          ((_Float16*)outp)[(((size_t)bb * H_ + hh) * L_ + ll) * DH_ + dd] =
              (_Float16)acc;
        } else if constexpr (EPI == 4) {  // V: [B,H,DH,L] (transposed)
          const int bb = row >> 10, ll = row & (L_ - 1);
          const int hh = col >> 6, dd = col & 63;
          ((_Float16*)outp)[(((size_t)bb * H_ + hh) * DH_ + dd) * L_ + ll] =
              (_Float16)acc;
        } else if constexpr (EPI == 1 || EPI == 3) {
          const size_t o = (size_t)row * N + col;
          ((float*)outp)[o] = acc + resid[o];
        } else {
          const size_t o = (size_t)row * N + col;
          ((_Float16*)outp)[o] = (_Float16)fmaxf(acc, 0.0f);
        }
      }
}

// ---------------------------------------------------------------------------
// Flash attention: one block = 128 query rows of one (b,h); 8 waves x 16 rows.
// 32-key chunks; K tile [key][d] and pre-transposed V tile [d][key] both
// streamed by the Tensor Data Mover, double-buffered.  A static all-ones LDS
// block makes the PV WMMA also produce the softmax row-sums.
// ---------------------------------------------------------------------------
__global__ __launch_bounds__(256) void attn_kernel(
    const _Float16* __restrict__ q, const _Float16* __restrict__ k,
    const _Float16* __restrict__ vt, _Float16* __restrict__ cc) {
  constexpr int KS = 72;  // Ks row stride (144B)
  constexpr int VS = 40;  // Vt row stride (80B)
  constexpr int PS = 40;  // P scratch row stride
  __shared__ _Float16 Ks[2][32 * KS];   // K chunk [key][d]
  __shared__ _Float16 Vt[2][64 * VS];   // V chunk [d][key] (global pre-transposed)
  __shared__ _Float16 Vones[16 * VS];   // all-ones rows -> row-sum columns
  __shared__ _Float16 Ps[8 * 16 * PS];  // per-wave P re-layout scratch

  const int tid  = threadIdx.x;
  const int wave = tid >> 5, lane = tid & 31;
  const int l15  = lane & 15, lh = lane >> 4;
  const int bh   = blockIdx.y;
  const int b    = bh / H_, h = bh % H_;
  const int m0   = blockIdx.x * 128;
  const size_t base = (size_t)bh * L_ * DH_;
  const _Float16* kptr = k + base;    // [L][DH] rows
  const _Float16* vtp  = vt + base;   // [DH][L] rows
  const float SC = 0.125f;            // 1/sqrt(DH)

  for (int i = tid; i < 16 * VS; i += 256)
    Vones[i] = (_Float16)(((i % VS) < 32) ? 1.0f : 0.0f);

  V16 qa[2];
  const int qrow = m0 + wave * 16 + l15;
#pragma unroll
  for (int kd = 0; kd < 2; ++kd) {
    const _Float16* p = q + base + (size_t)qrow * DH_ + kd * 32 + 8 * lh;
    qa[kd].h[0] = *(const v8h*)p;
    qa[kd].h[1] = *(const v8h*)(p + 16);
  }

  const v8f vzero = {0.f, 0.f, 0.f, 0.f, 0.f, 0.f, 0.f, 0.f};
  float mrow[8];
  v8f y[5];  // y[0..3] = output d-groups, y[4] = running softmax row-sum
#pragma unroll
  for (int r = 0; r < 8; ++r) mrow[r] = -3.0e38f;
#pragma unroll
  for (int g = 0; g < 5; ++g) y[g] = vzero;

#if USE_TDM
  const unsigned ksOff[2] = {(unsigned)(size_t)(const void*)&Ks[0][0],
                             (unsigned)(size_t)(const void*)&Ks[1][0]};
  const unsigned vtOff[2] = {(unsigned)(size_t)(const void*)&Vt[0][0],
                             (unsigned)(size_t)(const void*)&Vt[1][0]};
  if (wave == 0) {  // prologue: chunk 0
    tdm_load_2d(ksOff[0], kptr, 16, 32, 16, 4, 3);                    // 128B rows
    tdm_load_2d(vtOff[0], vtp, 8, 64, (unsigned)(L_ >> 2), 3, 3);     // 64B rows
  }
#endif

  const int nchunk = L_ / 32;
  for (int ci = 0; ci < nchunk; ++ci) {
    const int buf = ci & 1;
    __syncthreads();  // all waves done reading buf^1 (chunk ci-1)
#if USE_TDM
    if (wave == 0) {
      if (ci + 1 < nchunk) {  // overlap chunk ci+1 DMA with chunk ci compute
        const int kb1 = (ci + 1) * 32;
        tdm_load_2d(ksOff[buf ^ 1], kptr + (size_t)kb1 * DH_, 16, 32, 16, 4, 3);
        tdm_load_2d(vtOff[buf ^ 1], vtp + kb1, 8, 64, (unsigned)(L_ >> 2), 3, 3);
        __builtin_amdgcn_s_wait_tensorcnt(2);  // oldest two (chunk ci) done
      } else {
        __builtin_amdgcn_s_wait_tensorcnt(0);
      }
    }
#else
    {  // cooperative fallback: both tiles fully coalesced b128 copies
      const int kb = ci * 32;
      const int key = tid >> 3, dc = (tid & 7) * 8;
      *(v8h*)&Ks[buf][key * KS + dc] =
          *(const v8h*)(kptr + (size_t)(kb + key) * DH_ + dc);
      const int d = tid >> 2, kc = (tid & 3) * 8;
      *(v8h*)&Vt[buf][d * VS + kc] =
          *(const v8h*)(vtp + (size_t)d * L_ + kb + kc);
    }
#endif
    __syncthreads();  // chunk ci tiles visible

    // S = q . k^T : two 16x16 f32 frags (keys 0-15, 16-31)
    v8f c0 = vzero, c1 = vzero;
#pragma unroll
    for (int kd = 0; kd < 2; ++kd) {
      V16 b0, b1;
      const _Float16* p0 = &Ks[buf][l15 * KS + kd * 32 + 8 * lh];
      b0.h[0] = *(const v8h*)p0;
      b0.h[1] = *(const v8h*)(p0 + 16);
      const _Float16* p1 = &Ks[buf][(16 + l15) * KS + kd * 32 + 8 * lh];
      b1.h[0] = *(const v8h*)p1;
      b1.h[1] = *(const v8h*)(p1 + 16);
      c0 = wmma_f16(qa[kd].v, b0.v, c0);
      c1 = wmma_f16(qa[kd].v, b1.v, c1);
    }

    // Online softmax: VALU permlane row-max; row-sum comes from y[4] WMMA.
#pragma unroll
    for (int r = 0; r < 8; ++r) {
      const float vmax = rowmax16(fmaxf(c0[r], c1[r]));
      const float mn   = fmaxf(mrow[r], vmax);
      const float corr = __expf((mrow[r] - mn) * SC);
      const float p0   = __expf((c0[r] - mn) * SC);
      const float p1   = __expf((c1[r] - mn) * SC);
      mrow[r] = mn;
      y[0][r] *= corr; y[1][r] *= corr; y[2][r] *= corr;
      y[3][r] *= corr; y[4][r] *= corr;
      _Float16* pp = &Ps[wave * (16 * PS) + (r + 8 * lh) * PS];
      pp[l15]      = (_Float16)p0;
      pp[16 + l15] = (_Float16)p1;
    }
    // wave-internal DS RAW ordering before re-reading Ps
    asm volatile("s_wait_dscnt 0" ::: "memory");

    V16 pa;  // P as A-fragment (16 rows x 32 keys)
    const _Float16* pr = &Ps[wave * (16 * PS) + l15 * PS + 8 * lh];
    pa.h[0] = *(const v8h*)pr;
    pa.h[1] = *(const v8h*)(pr + 16);

#pragma unroll
    for (int ng = 0; ng < 5; ++ng) {  // ng==4 hits the all-ones rows -> rowsums
      V16 vbf;
      const _Float16* vp = (ng < 4)
          ? &Vt[buf][(ng * 16 + l15) * VS + 8 * lh]
          : &Vones[l15 * VS + 8 * lh];
      vbf.h[0] = *(const v8h*)vp;
      vbf.h[1] = *(const v8h*)(vp + 16);
      y[ng] = wmma_f16(pa.v, vbf.v, y[ng]);
    }
  }

  // Normalize (every lane of y[4] holds its row's sum) and write concat fp16
#pragma unroll
  for (int r = 0; r < 8; ++r) {
    const float inv = 1.0f / y[4][r];
    const int row = m0 + wave * 16 + r + 8 * lh;
    const size_t o = ((size_t)b * L_ + row) * E_ + h * DH_;
#pragma unroll
    for (int ng = 0; ng < 4; ++ng)
      cc[o + ng * 16 + l15] = (_Float16)(y[ng][r] * inv);
  }
}

// ---------------------------------------------------------------------------
// LayerNorm over E=1024. One row per block, 256 threads x 4 elements.
// ---------------------------------------------------------------------------
template <bool WB>
__global__ __launch_bounds__(256) void ln_kernel(
    const float* __restrict__ in, const float* __restrict__ gamma,
    const float* __restrict__ beta, float* __restrict__ outf,
    _Float16* __restrict__ outh) {
  __shared__ float red[256];
  const int tid = threadIdx.x;
  const size_t row = blockIdx.x;
  const float* xr = in + row * E_;

  float vals[4];
  float s = 0.f;
#pragma unroll
  for (int i = 0; i < 4; ++i) { vals[i] = xr[tid + i * 256]; s += vals[i]; }
  red[tid] = s;
  __syncthreads();
  for (int st = 128; st > 0; st >>= 1) {
    if (tid < st) red[tid] += red[tid + st];
    __syncthreads();
  }
  const float mean = red[0] * (1.0f / E_);
  __syncthreads();

  float vv = 0.f;
#pragma unroll
  for (int i = 0; i < 4; ++i) { const float d = vals[i] - mean; vv += d * d; }
  red[tid] = vv;
  __syncthreads();
  for (int st = 128; st > 0; st >>= 1) {
    if (tid < st) red[tid] += red[tid + st];
    __syncthreads();
  }
  const float rstd = rsqrtf(red[0] * (1.0f / E_) + 1e-10f);

#pragma unroll
  for (int i = 0; i < 4; ++i) {
    const int cidx = tid + i * 256;
    const float o = gamma[cidx] * (vals[i] - mean) * rstd + beta[cidx];
    outf[row * E_ + cidx] = o;
    if constexpr (WB) outh[row * E_ + cidx] = (_Float16)o;
  }
}

// ---------------------------------------------------------------------------
// Orchestration
// ---------------------------------------------------------------------------
extern "C" void kernel_launch(void* const* d_in, const int* in_sizes, int n_in,
                              void* d_out, int out_size, void* d_ws,
                              size_t ws_size, hipStream_t stream) {
  const float* x   = (const float*)d_in[0];
  const float* Wq  = (const float*)d_in[1];
  const float* bq  = (const float*)d_in[2];
  const float* Wk  = (const float*)d_in[3];
  const float* bk  = (const float*)d_in[4];
  const float* Wv  = (const float*)d_in[5];
  const float* bv  = (const float*)d_in[6];
  const float* Wo  = (const float*)d_in[7];
  const float* bo  = (const float*)d_in[8];
  const float* W1  = (const float*)d_in[9];
  const float* b1  = (const float*)d_in[10];
  const float* W2  = (const float*)d_in[11];
  const float* b2  = (const float*)d_in[12];
  const float* g1  = (const float*)d_in[13];
  const float* be1 = (const float*)d_in[14];
  const float* g2  = (const float*)d_in[15];
  const float* be2 = (const float*)d_in[16];
  float* out = (float*)d_out;

  const int NX  = B_ * L_ * E_;   // 8388608
  const int NWQ = H_ * E_ * DH_;  // 1048576
  const int NWO = E_ * E_;
  const int NW1 = E_ * FF_;       // 4194304
  const int M   = B_ * L_;        // 8192

  // ---- workspace carve (256B aligned) ----
  char* p = (char*)d_ws;
  auto nxt = [&](size_t bytes) -> char* {
    char* r = p;
    p += (bytes + 255) & ~(size_t)255;
    return r;
  };
  _Float16* xb   = (_Float16*)nxt((size_t)NX * 2);
  _Float16* wqb  = (_Float16*)nxt((size_t)NWQ * 2);  // transposed [H*DH, E]
  _Float16* wkb  = (_Float16*)nxt((size_t)NWQ * 2);
  _Float16* wvb  = (_Float16*)nxt((size_t)NWQ * 2);
  _Float16* wob  = (_Float16*)nxt((size_t)NWO * 2);  // transposed [E, E]
  _Float16* w1b  = (_Float16*)nxt((size_t)NW1 * 2);  // transposed [FF, E]
  _Float16* w2b  = (_Float16*)nxt((size_t)NW1 * 2);  // transposed [E, FF]
  _Float16* qb   = (_Float16*)nxt((size_t)NX * 2);   // [B,H,L,DH]
  _Float16* kb   = (_Float16*)nxt((size_t)NX * 2);   // [B,H,L,DH]
  _Float16* vb   = (_Float16*)nxt((size_t)NX * 2);   // [B,H,DH,L] (transposed)
  _Float16* ccb  = (_Float16*)nxt((size_t)NX * 2);   // concat fp16
  float*    out1 = (float*)nxt((size_t)NX * 4);
  float*    o2f  = (float*)nxt((size_t)NX * 4);
  _Float16* o2h  = (_Float16*)nxt((size_t)NX * 2);
  _Float16* hb = qb;    // FFN hidden (64MB) reuses dead qb..ccb
  float*    t2 = out1;  // second residual reuses dead out1

  // ---- conversions; all weights pre-transposed to [N,K] fp16 ----
  cvt_kernel<<<(NX + 255) / 256, 256, 0, stream>>>(x, xb, NX);
  {
    dim3 gt((E_ * DH_ + 255) / 256, H_);  // per-head [E,DH] -> [DH,E]
    cvt_tr_kernel<<<gt, 256, 0, stream>>>(Wq, wqb, E_, DH_);
    cvt_tr_kernel<<<gt, 256, 0, stream>>>(Wk, wkb, E_, DH_);
    cvt_tr_kernel<<<gt, 256, 0, stream>>>(Wv, wvb, E_, DH_);
  }
  cvt_tr_kernel<<<dim3((NWO + 255) / 256, 1), 256, 0, stream>>>(Wo, wob, E_, E_);
  cvt_tr_kernel<<<dim3((NW1 + 255) / 256, 1), 256, 0, stream>>>(W1, w1b, E_, FF_);
  cvt_tr_kernel<<<dim3((NW1 + 255) / 256, 1), 256, 0, stream>>>(W2, w2b, FF_, E_);

  // ---- QKV projections (V written pre-transposed for the attention TDM) ----
  dim3 gq(E_ / 128, M / 128);  // (8, 64)
  gemm_kernel<0><<<gq, 256, 0, stream>>>(xb, E_, E_, wqb, E_, bq, nullptr, qb);
  gemm_kernel<0><<<gq, 256, 0, stream>>>(xb, E_, E_, wkb, E_, bk, nullptr, kb);
  gemm_kernel<4><<<gq, 256, 0, stream>>>(xb, E_, E_, wvb, E_, bv, nullptr, vb);

  // ---- flash attention -> concat fp16 ----
  dim3 ga(L_ / 128, B_ * H_);  // (8, 128)
  attn_kernel<<<ga, 256, 0, stream>>>(qb, kb, vb, ccb);

  // ---- out projection + bias + residual(x) ----
  gemm_kernel<1><<<gq, 256, 0, stream>>>(ccb, E_, E_, wob, E_, bo, x, out1);

  // ---- LN1 ----
  ln_kernel<true><<<M, 256, 0, stream>>>(out1, g1, be1, o2f, o2h);

  // ---- FFN ----
  dim3 gf1(FF_ / 128, M / 128);  // (32, 64)
  gemm_kernel<2><<<gf1, 256, 0, stream>>>(o2h, E_, E_, w1b, FF_, b1, nullptr, hb);
  gemm_kernel<3><<<gq, 256, 0, stream>>>(hb, FF_, FF_, w2b, E_, b2, o2f, t2);

  // ---- LN2 -> final output ----
  ln_kernel<false><<<M, 256, 0, stream>>>(t2, g2, be2, out, nullptr);
}
